// PyTorchConv2dAutograd_67336497267120
// MI455X (gfx1250) — matrix operand order repaired and checked by reading, not compiled, not run on hardware
//
#include <hip/hip_runtime.h>

typedef __attribute__((ext_vector_type(16))) _Float16 v16h;
typedef __attribute__((ext_vector_type(8)))  float    v8f;

// Problem constants (from the reference)
#define N_IMG 32
#define C_IN  64
#define H_IN  64
#define W_IN  64
#define F_OUT 64
#define OH    62
#define OW    62
#define PIX_PER_IMG (OH * OW)             // 3844
#define M_TOTAL     (N_IMG * PIX_PER_IMG) // 123008
#define TILE_M 128
#define KSTEP  32
#define NKITER 18                          // 9 (r,s) pairs * 2 c-halves; K = 576

#define A_STRIDE 40   // halves per A row (32 data + 8 pad) -> 80B, 16B-aligned rows
#define B_STRIDE 40   // halves per B row
#define EP_STRIDE 65  // floats per epilogue row (odd -> no LDS bank conflicts)

#define SMEM_STAGE ((2 * TILE_M * A_STRIDE + 2 * F_OUT * B_STRIDE) * 2) // 30720 B
#define SMEM_EPIL  (TILE_M * EP_STRIDE * 4)                             // 33280 B
#define SMEM_BYTES (SMEM_EPIL > SMEM_STAGE ? SMEM_EPIL : SMEM_STAGE)

// K-ordering: k' = (r*3+s)*C + c  (A and B both use it, so any order is valid).
// K-step kt: rs = kt>>1 (fixed r,s), cbase = (kt&1)*32, klocal = c - cbase.

__launch_bounds__(256, 2)
__global__ void conv2d_implicit_gemm_wmma(const float* __restrict__ x,
                                          const float* __restrict__ w,
                                          const float* __restrict__ b,
                                          float* __restrict__ out) {
  __shared__ __align__(16) char smem[SMEM_BYTES];
  _Float16* As = (_Float16*)smem;                                   // 2 bufs * 128 * 40
  _Float16* Bs = (_Float16*)(smem + 2 * TILE_M * A_STRIDE * 2);     // 2 bufs * 64 * 40
  float*    Ep = (float*)smem;                                      // reused post-loop

  const int t    = threadIdx.x;
  const int wave = t >> 5;
  const int lane = t & 31;
  const int rowl = lane & 15;   // M/N position inside a 16-wide fragment
  const int hi   = lane >> 4;   // lane-group select per ISA fragment layout

  // One output pixel per thread (mod 128) for staging + epilogue.
  const int p     = t & 127;
  const int m     = blockIdx.x * TILE_M + p;
  const int n_img = m / PIX_PER_IMG;
  const int rm    = m % PIX_PER_IMG;
  const int oh    = rm / OW;
  const int ow    = rm % OW;
  const float* xbase = x + ((size_t)(n_img * C_IN) * H_IN + oh) * W_IN + ow;
  const int khalf = t >> 7; // 0 or 1: klocal parity this thread stages

  // Per-thread constant offsets for B staging (f = i>>5, kl = i&31).
  int b_goff[8]; // element offset into w: f*576 + kl*9  (+ uniform cbase*9+rs)
  int b_loff[8]; // LDS halves offset: f*B_STRIDE + kl
#pragma unroll
  for (int j = 0; j < 8; ++j) {
    const int i = j * 256 + t;
    b_goff[j] = (i >> 5) * (C_IN * 9) + (i & 31) * 9;
    b_loff[j] = (i >> 5) * B_STRIDE + (i & 31);
  }

  // Phase 1: issue all global loads for K-step `kt` (no waits; loads stay in
  // flight while WMMAs on the other LDS buffer execute).
  float a_reg[16];
  float b_reg[8];
  auto load_globals = [&](int kt) {
    const int rs = kt >> 1;           // uniform (SALU)
    const int r  = rs / 3;
    const int s  = rs - 3 * r;
    const int cb = (kt & 1) << 5;
    // A: c = cb + 2j + khalf ; addr = xbase + c*H*W + r*W + s
    const float* xa = xbase + (cb + khalf) * (H_IN * W_IN) + r * W_IN + s;
#pragma unroll
    for (int j = 0; j < 16; ++j)
      a_reg[j] = xa[j * 2 * H_IN * W_IN]; // j*32KB folds into load immediate
    // B: w[f*576 + (cb+kl)*9 + rs]
    const int soff = cb * 9 + rs;       // uniform (SALU)
#pragma unroll
    for (int j = 0; j < 8; ++j)
      b_reg[j] = w[b_goff[j] + soff];
  };

  // Phase 2: convert f32->f16 and commit to LDS staging buffer `buf`.
  auto store_stage = [&](int buf) {
    _Float16* Ab = As + buf * TILE_M * A_STRIDE + p * A_STRIDE + khalf;
#pragma unroll
    for (int j = 0; j < 16; ++j)
      Ab[2 * j] = (_Float16)a_reg[j];
    _Float16* Bb = Bs + buf * F_OUT * B_STRIDE;
#pragma unroll
    for (int j = 0; j < 8; ++j)
      Bb[b_loff[j]] = (_Float16)b_reg[j];
  };

  v8f acc[4] = {};

  load_globals(0);
  store_stage(0);
  __syncthreads();

  for (int kt = 0; kt < NKITER; ++kt) {
    const int buf = kt & 1;
    if (kt + 1 < NKITER) load_globals(kt + 1); // issue next tile's loads

    const _Float16* Ab = As + buf * TILE_M * A_STRIDE;
    const _Float16* Bb = Bs + buf * F_OUT * B_STRIDE;

    // A fragment: 16x32 f16 (ISA 7.12.2): lanes 0-15 K=0..7/16..23,
    // lanes 16-31 K=8..15/24..31; two 16B chunks per lane.
    union { float4 q[2]; v16h v; } af;
    const _Float16* arow = Ab + (wave * 16 + rowl) * A_STRIDE + hi * 8;
    af.q[0] = *(const float4*)(arow);        // K = hi*8 + 0..7
    af.q[1] = *(const float4*)(arow + 16);   // K = 16 + hi*8 + 0..7

#pragma unroll
    for (int ft = 0; ft < 4; ++ft) {
      // B fragment: 32x16 f16: lanes 0-15 K=0..7/16..23, lanes 16-31 K=8..15/24..31.
      union { float4 q[2]; v16h v; } bf;
      const _Float16* brow = Bb + (ft * 16 + rowl) * B_STRIDE + hi * 8;
      bf.q[0] = *(const float4*)(brow);
      bf.q[1] = *(const float4*)(brow + 16);
      acc[ft] = __builtin_amdgcn_wmma_f32_16x16x32_f16(
          false, af.v, false, bf.v, (short)0, acc[ft], false, false);
    }

    if (kt + 1 < NKITER) store_stage(buf ^ 1); // single wait, convert, commit
    __syncthreads(); // stage writes + fragment reads done before buffer reuse
  }

  // Epilogue: D-fragment layout -> LDS [pixel][f], then coalesced global stores.
#pragma unroll
  for (int ft = 0; ft < 4; ++ft) {
#pragma unroll
    for (int r = 0; r < 8; ++r) {
      const int ml = wave * 16 + hi * 8 + r; // M row per C/D VGPR layout
      const int f  = ft * 16 + rowl;         // N col = lane%16
      Ep[ml * EP_STRIDE + f] = acc[ft][r];
    }
  }
  __syncthreads();

  float* obase = out + ((size_t)(n_img * F_OUT) * OH + oh) * OW + ow;
#pragma unroll
  for (int j = 0; j < 32; ++j) {
    const int f = 2 * j + khalf;
    obase[(size_t)f * PIX_PER_IMG] = Ep[p * EP_STRIDE + f] + b[f];
  }
}

extern "C" void kernel_launch(void* const* d_in, const int* in_sizes, int n_in,
                              void* d_out, int out_size, void* d_ws, size_t ws_size,
                              hipStream_t stream) {
  (void)in_sizes; (void)n_in; (void)out_size; (void)d_ws; (void)ws_size;
  const float* x = (const float*)d_in[0];
  const float* w = (const float*)d_in[1];
  const float* b = (const float*)d_in[2];
  float* out = (float*)d_out;
  const int grid = M_TOTAL / TILE_M; // 961 exact
  conv2d_implicit_gemm_wmma<<<grid, 256, 0, stream>>>(x, w, b, out);
}
